// ResonanceAttention_60979945668676
// MI455X (gfx1250) — matrix operand order repaired and checked by reading, not compiled, not run on hardware
//
#include <hip/hip_runtime.h>

typedef __bf16 bf16;
typedef __attribute__((ext_vector_type(16))) __bf16 v16bf;
typedef __attribute__((ext_vector_type(8)))  __bf16 v8bf;
typedef __attribute__((ext_vector_type(4)))  __bf16 v4bf;
typedef __attribute__((ext_vector_type(8)))  float  v8f;
typedef __attribute__((ext_vector_type(4)))  float  v4f;

#define B_  4
#define S_  2048
#define D_  1024
#define H_  16
#define HD_ 64

__device__ __forceinline__ bf16 f2bf(float f) {
  unsigned u = __builtin_bit_cast(unsigned, f);
  u += 0x7fffu + ((u >> 16) & 1u);                 // round-to-nearest-even
  unsigned short h = (unsigned short)(u >> 16);
  return __builtin_bit_cast(bf16, h);
}

__device__ __forceinline__ v16bf ld_frag(const bf16* p0, const bf16* p1) {
  union { v16bf v; v8bf h[2]; } u;
  u.h[0] = *(const v8bf*)p0;
  u.h[1] = *(const v8bf*)p1;
  return u.v;
}

__device__ __forceinline__ v8f wmma_bf16(v16bf a, v16bf b, v8f c) {
  return __builtin_amdgcn_wmma_f32_16x16x32_bf16(false, a, false, b,
                                                 (short)0, c, false, false);
}

// Async memory->LDS copy, 16B per lane, GVS addressing (SGPR base + VGPR off).
// Tracked by ASYNCcnt; paired with s_wait_asynccnt for double buffering.
__device__ __forceinline__ void async_b128(const bf16* base, unsigned byte_off,
                                           bf16* lds_dst) {
  unsigned lds = (unsigned)(uintptr_t)lds_dst;   // low 32 bits = LDS offset
  asm volatile("global_load_async_to_lds_b128 %0, %1, %2"
               :: "v"(lds), "v"(byte_off),
                  "s"((unsigned long long)(uintptr_t)base)
               : "memory");
}

__device__ __forceinline__ void wait_async(int pending) {
  if (pending) asm volatile("s_wait_asynccnt 0x3" ::: "memory");
  else         asm volatile("s_wait_asynccnt 0x0" ::: "memory");
}
__device__ __forceinline__ void wait_async2(int pending) {
  if (pending) asm volatile("s_wait_asynccnt 0x2" ::: "memory");
  else         asm volatile("s_wait_asynccnt 0x0" ::: "memory");
}

// ---------------------------------------------------------------------------
// Kernel 0: one-shot f32 -> bf16 conversion (x and weights), so that GEMM
// staging becomes a pure async copy with zero VALU work per tile.
// ---------------------------------------------------------------------------
__global__ __launch_bounds__(256)
void to_bf16(const float* __restrict__ in, bf16* __restrict__ outp, int n) {
  int i = (blockIdx.x * 256 + threadIdx.x) * 4;
  if (i < n) {
    v4f v = *(const v4f*)(in + i);
    v4bf o;
    for (int j = 0; j < 4; ++j) o[j] = f2bf(v[j]);
    *(v4bf*)(outp + i) = o;
  }
}

// ---------------------------------------------------------------------------
// Kernel 1: q/k/v projections from bf16 x and bf16 W. Block tile 128x64,
// 8 waves (4x2), wave tile 32x32 (2x2 WMMA). Double-buffered async staging.
// q is pre-scaled by 1/sqrt(hd). v is stored TRANSPOSED [B,H,hd,S] so the
// attention kernel can async-copy V tiles directly.
// ---------------------------------------------------------------------------
__global__ __launch_bounds__(256)
void qkv_gemm(const bf16* __restrict__ xb,
              const bf16* __restrict__ Wq, const float* __restrict__ bq,
              const bf16* __restrict__ Wk, const float* __restrict__ bk,
              const bf16* __restrict__ Wv, const float* __restrict__ bv,
              bf16* __restrict__ qo, bf16* __restrict__ ko, bf16* __restrict__ vo)
{
  __shared__ bf16 xs[2][128][48];   // row stride 96B (16B multiple)
  __shared__ bf16 ws[2][64][48];

  const int z = blockIdx.z;
  const bf16*  W    = (z == 0) ? Wq : (z == 1) ? Wk : Wv;
  const float* bias = (z == 0) ? bq : (z == 1) ? bk : bv;
  bf16* outp        = (z == 0) ? qo : (z == 1) ? ko : vo;
  const float scale = (z == 0) ? 0.125f : 1.0f;

  const int M0 = blockIdx.x * 128;
  const int N0 = blockIdx.y * 64;
  const int t = threadIdx.x;
  const int w = t >> 5, lane = t & 31;
  const int lh = lane & 15, hi = lane >> 4;
  const int wm = w & 3, wn = w >> 2;
  const int c0 = hi ? 8 : 0;

  // staging decomposition: x tile 128x32 = 512 16B chunks (2/thread),
  // W tile 64x32 = 256 chunks (1/thread)
  const int xrow0 = t >> 2;             // 0..63
  const int xcol  = (t & 3) * 8;        // 0,8,16,24

  v8f acc[2][2] = {};

#define QKV_STAGE(K0, BUF)                                                     \
  {                                                                            \
    async_b128(xb, (unsigned)(((M0 + xrow0)      * D_ + (K0) + xcol) * 2),     \
               &xs[BUF][xrow0][xcol]);                                         \
    async_b128(xb, (unsigned)(((M0 + xrow0 + 64) * D_ + (K0) + xcol) * 2),     \
               &xs[BUF][xrow0 + 64][xcol]);                                    \
    async_b128(W,  (unsigned)(((N0 + xrow0)      * D_ + (K0) + xcol) * 2),     \
               &ws[BUF][xrow0][xcol]);                                         \
  }

  QKV_STAGE(0, 0);
  for (int kt = 0; kt < D_ / 32; ++kt) {
    const int cur = kt & 1;
    if (kt + 1 < D_ / 32) { QKV_STAGE((kt + 1) * 32, cur ^ 1); wait_async(1); }
    else                  { wait_async(0); }
    __syncthreads();

    v16bf a0 = ld_frag(&xs[cur][wm*32 +      lh][c0], &xs[cur][wm*32 +      lh][c0 + 16]);
    v16bf a1 = ld_frag(&xs[cur][wm*32 + 16 + lh][c0], &xs[cur][wm*32 + 16 + lh][c0 + 16]);
    v16bf b0 = ld_frag(&ws[cur][wn*32 +      lh][c0], &ws[cur][wn*32 +      lh][c0 + 16]);
    v16bf b1 = ld_frag(&ws[cur][wn*32 + 16 + lh][c0], &ws[cur][wn*32 + 16 + lh][c0 + 16]);
    acc[0][0] = wmma_bf16(a0, b0, acc[0][0]);
    acc[0][1] = wmma_bf16(a0, b1, acc[0][1]);
    acc[1][0] = wmma_bf16(a1, b0, acc[1][0]);
    acc[1][1] = wmma_bf16(a1, b1, acc[1][1]);
    __syncthreads();
  }
#undef QKV_STAGE

  for (int mi = 0; mi < 2; ++mi)
    for (int ni = 0; ni < 2; ++ni)
      for (int r = 0; r < 8; ++r) {
        int i = M0 + wm*32 + mi*16 + r + hi*8;
        int j = N0 + wn*32 + ni*16 + lh;
        float val = (acc[mi][ni][r] + bias[j]) * scale;
        int bb = i >> 11, ss = i & (S_ - 1);
        int hh = j >> 6,  dd = j & 63;
        size_t idx = (z == 2)
          ? ((size_t)(bb*H_ + hh) * HD_ + dd) * S_ + ss    // v transposed
          : ((size_t)(bb*H_ + hh) * S_ + ss) * HD_ + dd;   // q,k row-major
        outp[idx] = f2bf(val);
      }
}

// ---------------------------------------------------------------------------
// Kernel 2: flash attention + entropy. Block = (q-tile of 128, one (b,h)).
// K [S,hd] and V^T [hd,S] tiles async-copied into double-buffered LDS.
// Online softmax; entropy H = m + log(l) - e/l via atomics.
// ---------------------------------------------------------------------------
__global__ __launch_bounds__(256)
void flash_attn(const bf16* __restrict__ q, const bf16* __restrict__ k,
                const bf16* __restrict__ vT, const float* __restrict__ rb,
                bf16* __restrict__ ctx, float* __restrict__ ent_acc)
{
  __shared__ bf16 ks[2][32][72];     // [j][d], stride 144B
  __shared__ bf16 vs[2][64][48];     // [d][j] (already transposed), stride 96B
  __shared__ bf16 ps[8][16][48];     // per-wave P tile [row][j]

  const int qt = blockIdx.x;
  const int bh = blockIdx.y;
  const int b = bh >> 4, h = bh & 15;
  const int t = threadIdx.x;
  const int w = t >> 5, lane = t & 31;
  const int lh = lane & 15, hi = lane >> 4;
  const int c0 = hi ? 8 : 0;

  const size_t head = (size_t)bh * S_ * HD_;
  const bf16* qh = q  + head;
  const bf16* kh = k  + head;
  const bf16* vh = vT + head;        // [hd][S]
  const float rbv = rb[h];

  const int qbase = qt * 128 + w * 16;
  const bf16* qrow = qh + (size_t)(qbase + lh) * HD_;
  v16bf qa0 = ld_frag(qrow + c0,      qrow + c0 + 16);
  v16bf qa1 = ld_frag(qrow + 32 + c0, qrow + 32 + c0 + 16);

  float m[8], l[8], e[8];
  v8f acc[4] = {};
  for (int r = 0; r < 8; ++r) { m[r] = -1e30f; l[r] = 0.f; e[r] = 0.f; }

  // staging: K tile 32x64 = 256 chunks (1/thread); V^T tile 64x32 = 256 chunks
  const int krow = t >> 3, kcol = (t & 7) * 8;   // K: row 0..31, col 0..56
  const int vrow = t >> 2, vcol = (t & 3) * 8;   // V^T: d 0..63, j 0..24

#define ATT_STAGE(J0, BUF)                                                     \
  {                                                                            \
    async_b128(kh, (unsigned)((((J0) + krow) * HD_ + kcol) * 2),               \
               &ks[BUF][krow][kcol]);                                          \
    async_b128(vh, (unsigned)((vrow * S_ + (J0) + vcol) * 2),                  \
               &vs[BUF][vrow][vcol]);                                          \
  }

  ATT_STAGE(0, 0);
  for (int jt = 0; jt < S_ / 32; ++jt) {
    const int cur = jt & 1;
    if (jt + 1 < S_ / 32) { ATT_STAGE((jt + 1) * 32, cur ^ 1); wait_async2(1); }
    else                  { wait_async2(0); }
    __syncthreads();

    // scores: S-tile 16x32 = two 16x16 C frags, K=64 in two WMMA steps
    v8f sc0 = {}, sc1 = {};
    {
      v16bf bk0a = ld_frag(&ks[cur][lh][c0],      &ks[cur][lh][c0 + 16]);
      v16bf bk0b = ld_frag(&ks[cur][lh][32 + c0], &ks[cur][lh][32 + c0 + 16]);
      sc0 = wmma_bf16(qa0, bk0a, sc0);
      sc0 = wmma_bf16(qa1, bk0b, sc0);
      v16bf bk1a = ld_frag(&ks[cur][16 + lh][c0],      &ks[cur][16 + lh][c0 + 16]);
      v16bf bk1b = ld_frag(&ks[cur][16 + lh][32 + c0], &ks[cur][16 + lh][32 + c0 + 16]);
      sc1 = wmma_bf16(qa0, bk1a, sc1);
      sc1 = wmma_bf16(qa1, bk1b, sc1);
    }

    float p0[8], p1[8];
    for (int r = 0; r < 8; ++r) {
      float s0 = sc0[r] + rbv, s1 = sc1[r] + rbv;
      float tm = fmaxf(s0, s1);
      tm = fmaxf(tm, __shfl_xor(tm, 1, 32));
      tm = fmaxf(tm, __shfl_xor(tm, 2, 32));
      tm = fmaxf(tm, __shfl_xor(tm, 4, 32));
      tm = fmaxf(tm, __shfl_xor(tm, 8, 32));
      float mn = fmaxf(m[r], tm);
      p0[r] = __expf(s0 - mn);
      p1[r] = __expf(s1 - mn);
      float ts = p0[r] + p1[r];
      float te = p0[r] * s0 + p1[r] * s1;
      ts += __shfl_xor(ts, 1, 32);  te += __shfl_xor(te, 1, 32);
      ts += __shfl_xor(ts, 2, 32);  te += __shfl_xor(te, 2, 32);
      ts += __shfl_xor(ts, 4, 32);  te += __shfl_xor(te, 4, 32);
      ts += __shfl_xor(ts, 8, 32);  te += __shfl_xor(te, 8, 32);
      float corr = __expf(m[r] - mn);
      l[r] = l[r] * corr + ts;
      e[r] = e[r] * corr + te;
      m[r] = mn;
      for (int f = 0; f < 4; ++f) acc[f][r] *= corr;
    }

    // C-layout -> A-layout pivot through LDS (wave-private buffer)
    for (int r = 0; r < 8; ++r) {
      ps[w][r + hi*8][lh]      = f2bf(p0[r]);
      ps[w][r + hi*8][16 + lh] = f2bf(p1[r]);
    }
    asm volatile("s_wait_dscnt 0x0" ::: "memory");

    v16bf pa = ld_frag(&ps[w][lh][c0], &ps[w][lh][c0 + 16]);
    for (int f = 0; f < 4; ++f) {
      v16bf bv_ = ld_frag(&vs[cur][f*16 + lh][c0], &vs[cur][f*16 + lh][c0 + 16]);
      acc[f] = wmma_bf16(pa, bv_, acc[f]);
    }
    __syncthreads();
  }
#undef ATT_STAGE

  float ent = 0.f;
  float inv[8];
  for (int r = 0; r < 8; ++r) {
    inv[r] = 1.0f / l[r];
    ent += m[r] + __logf(l[r]) - e[r] * inv[r];
  }
  for (int f = 0; f < 4; ++f)
    for (int r = 0; r < 8; ++r) {
      int srow = qbase + r + hi*8;
      size_t idx = ((size_t)(b * S_ + srow)) * D_ + h*HD_ + f*16 + lh;
      ctx[idx] = f2bf(acc[f][r] * inv[r]);
    }
  if (lh == 0) atomicAdd(&ent_acc[bh], ent);   // lanes 0 (rows 0-7), 16 (rows 8-15)
}

// ---------------------------------------------------------------------------
// Kernel 3: out = ctx @ Wo^T + bo (f32 output), async double-buffered staging
// ---------------------------------------------------------------------------
__global__ __launch_bounds__(256)
void out_gemm(const bf16* __restrict__ ctx, const bf16* __restrict__ Wo,
              const float* __restrict__ bo, float* __restrict__ out)
{
  __shared__ bf16 xs[2][128][48];
  __shared__ bf16 ws[2][64][48];
  const int M0 = blockIdx.x * 128, N0 = blockIdx.y * 64;
  const int t = threadIdx.x, w = t >> 5, lane = t & 31;
  const int lh = lane & 15, hi = lane >> 4;
  const int wm = w & 3, wn = w >> 2;
  const int c0 = hi ? 8 : 0;
  const int xrow0 = t >> 2, xcol = (t & 3) * 8;

  v8f acc[2][2] = {};

#define OUT_STAGE(K0, BUF)                                                     \
  {                                                                            \
    async_b128(ctx, (unsigned)(((M0 + xrow0)      * D_ + (K0) + xcol) * 2),    \
               &xs[BUF][xrow0][xcol]);                                         \
    async_b128(ctx, (unsigned)(((M0 + xrow0 + 64) * D_ + (K0) + xcol) * 2),    \
               &xs[BUF][xrow0 + 64][xcol]);                                    \
    async_b128(Wo,  (unsigned)(((N0 + xrow0)      * D_ + (K0) + xcol) * 2),    \
               &ws[BUF][xrow0][xcol]);                                         \
  }

  OUT_STAGE(0, 0);
  for (int kt = 0; kt < D_ / 32; ++kt) {
    const int cur = kt & 1;
    if (kt + 1 < D_ / 32) { OUT_STAGE((kt + 1) * 32, cur ^ 1); wait_async(1); }
    else                  { wait_async(0); }
    __syncthreads();

    v16bf a0 = ld_frag(&xs[cur][wm*32 +      lh][c0], &xs[cur][wm*32 +      lh][c0 + 16]);
    v16bf a1 = ld_frag(&xs[cur][wm*32 + 16 + lh][c0], &xs[cur][wm*32 + 16 + lh][c0 + 16]);
    v16bf b0 = ld_frag(&ws[cur][wn*32 +      lh][c0], &ws[cur][wn*32 +      lh][c0 + 16]);
    v16bf b1 = ld_frag(&ws[cur][wn*32 + 16 + lh][c0], &ws[cur][wn*32 + 16 + lh][c0 + 16]);
    acc[0][0] = wmma_bf16(a0, b0, acc[0][0]);
    acc[0][1] = wmma_bf16(a0, b1, acc[0][1]);
    acc[1][0] = wmma_bf16(a1, b0, acc[1][0]);
    acc[1][1] = wmma_bf16(a1, b1, acc[1][1]);
    __syncthreads();
  }
#undef OUT_STAGE

  for (int mi = 0; mi < 2; ++mi)
    for (int ni = 0; ni < 2; ++ni)
      for (int r = 0; r < 8; ++r) {
        int i = M0 + wm*32 + mi*16 + r + hi*8;
        int j = N0 + wn*32 + ni*16 + lh;
        out[(size_t)i * D_ + j] = acc[mi][ni][r] + bo[j];
      }
}

__global__ void ent_finalize(const float* __restrict__ acc, float* __restrict__ out) {
  int i = threadIdx.x;
  if (i < B_ * H_) out[i] = acc[i] * (1.0f / (float)S_);
}

// ---------------------------------------------------------------------------
extern "C" void kernel_launch(void* const* d_in, const int* in_sizes, int n_in,
                              void* d_out, int out_size, void* d_ws, size_t ws_size,
                              hipStream_t stream) {
  const float* x  = (const float*)d_in[0];
  const float* Wq = (const float*)d_in[1];
  const float* bq = (const float*)d_in[2];
  const float* Wk = (const float*)d_in[3];
  const float* bk = (const float*)d_in[4];
  const float* Wv = (const float*)d_in[5];
  const float* bv = (const float*)d_in[6];
  const float* Wo = (const float*)d_in[7];
  const float* bo = (const float*)d_in[8];
  const float* rb = (const float*)d_in[9];

  char* ws = (char*)d_ws;
  const size_t x_bytes   = (size_t)B_ * S_ * D_ * sizeof(bf16);        // 16 MB
  const size_t w_bytes   = (size_t)D_ * D_ * sizeof(bf16);             //  2 MB
  const size_t qkv_bytes = (size_t)B_ * H_ * S_ * HD_ * sizeof(bf16);  // 16 MB

  bf16* xb  = (bf16*)(ws);
  bf16* wqb = (bf16*)(ws + x_bytes);
  bf16* wkb = (bf16*)(ws + x_bytes + 1 * w_bytes);
  bf16* wvb = (bf16*)(ws + x_bytes + 2 * w_bytes);
  bf16* wob = (bf16*)(ws + x_bytes + 3 * w_bytes);
  char* p2  = ws + x_bytes + 4 * w_bytes;
  bf16* qb  = (bf16*)(p2);
  bf16* kb  = (bf16*)(p2 + qkv_bytes);
  bf16* vb  = (bf16*)(p2 + 2 * qkv_bytes);   // stored transposed [B,H,hd,S]
  bf16* ctx = (bf16*)(p2 + 3 * qkv_bytes);
  float* ent = (float*)(p2 + 4 * qkv_bytes);

  float* out = (float*)d_out;
  float* ent_out = out + (size_t)B_ * S_ * D_;

  hipMemsetAsync(ent, 0, B_ * H_ * sizeof(float), stream);

  const int nx = B_ * S_ * D_, nw = D_ * D_;
  to_bf16<<<nx / 1024, 256, 0, stream>>>(x,  xb,  nx);
  to_bf16<<<nw / 1024, 256, 0, stream>>>(Wq, wqb, nw);
  to_bf16<<<nw / 1024, 256, 0, stream>>>(Wk, wkb, nw);
  to_bf16<<<nw / 1024, 256, 0, stream>>>(Wv, wvb, nw);
  to_bf16<<<nw / 1024, 256, 0, stream>>>(Wo, wob, nw);

  qkv_gemm<<<dim3(64, 16, 3), 256, 0, stream>>>(xb, wqb, bq, wkb, bk, wvb, bv,
                                                qb, kb, vb);
  flash_attn<<<dim3(S_ / 128, B_ * H_), 256, 0, stream>>>(qb, kb, vb, rb, ctx, ent);
  out_gemm<<<dim3(64, 16), 256, 0, stream>>>(ctx, wob, bo, out);
  ent_finalize<<<1, 64, 0, stream>>>(ent, ent_out);
}
